// BasicQuantumAttention_63093069578344
// MI455X (gfx1250) — matrix operand order repaired and checked by reading, not compiled, not run on hardware
//
#include <hip/hip_runtime.h>
#include <hip/hip_bf16.h>
#include <cstdint>

typedef __attribute__((ext_vector_type(16))) _Float16 v16h;
typedef __attribute__((ext_vector_type(8)))  _Float16 v8h;
typedef __attribute__((ext_vector_type(8)))  float    v8f;
typedef __attribute__((ext_vector_type(4)))  float    vf4;

union Frag16 { v16h v; v8h h[2]; };

#define LDH 40   // 32 K-halves + 8 pad halves per LDS row -> conflict-free 16B frag reads

// gfx1250 async global->LDS copy, 16B per lane, tracked by ASYNCcnt.
// (inline asm: portable across ROCm-7.2 / amdgpu-toolchain builtin arity skew)
__device__ __forceinline__ void async_load_b128(uint32_t lds_addr, const void* gptr) {
  asm volatile("global_load_async_to_lds_b128 %0, %1, off"
               :: "v"(lds_addr), "v"((uint64_t)(uintptr_t)gptr)
               : "memory");
}
__device__ __forceinline__ void wait_asynccnt0() {
  asm volatile("s_wait_asynccnt 0x0" ::: "memory");
}
__device__ __forceinline__ uint32_t lds_addr_of(const _Float16* p) {
  return (uint32_t)(uintptr_t)p;
}

// ---------------------------------------------------------------------------
// Generic f16 WMMA GEMM: C[m,n] = sum_k A[m,k] * B[n,k] (+ bias[n])
//   A: M x K row-major f16
//   B: if !B_KN: N x K row-major f16 (weights / K^T); if B_KN: stored K x N
//      (token-major V) and transposed into LDS.
//   Output: f16 (OUT_HALF) or f32; SPLIT writes f32 into two half-width dsts.
// Block tile 128x128, K-slab 32; 8 waves (2 M x 4 N), 4x2 WMMA accs per wave.
// Double-buffered LDS, async global->LDS staging overlapped with WMMA.
// ---------------------------------------------------------------------------
template<bool B_KN, bool OUT_HALF, bool SPLIT>
__global__ __launch_bounds__(256) void gemm_wmma(
    const _Float16* __restrict__ A, const _Float16* __restrict__ Bm,
    const float* __restrict__ bias,
    _Float16* __restrict__ Ch, float* __restrict__ Cf, float* __restrict__ Cf2,
    int K, int lda, int ldb, int ldc,
    long long sA, long long sB, long long sC, int splitN)
{
  __shared__ _Float16 As[2][128 * LDH];
  __shared__ _Float16 Bs[2][128 * LDH];

  const int tid  = threadIdx.x;
  const int lane = tid & 31;
  const int wave = tid >> 5;
  const int wm   = wave >> 2;   // 0..1  (M)
  const int wn   = wave & 3;    // 0..3  (N)
  const int bz   = blockIdx.z;
  const long long m0 = (long long)blockIdx.y * 128;
  const long long n0 = (long long)blockIdx.x * 128;

  A  += (long long)bz * sA;
  Bm += (long long)bz * sB;

  v8f zero = {};
  v8f acc[4][2];
#pragma unroll
  for (int i = 0; i < 4; ++i)
#pragma unroll
    for (int j = 0; j < 2; ++j) acc[i][j] = zero;

  // fragment addressing (gfx1250 wave32 WMMA VGPR layouts)
  const int arow = lane & 15;          // M (A) / N (B) index
  const int ako  = (lane >> 4) * 8;    // A: K 0..7 / 8..15 (+16 for hi half)
  const int bko  = (lane >> 4) * 8;    // B: K 0..7 / 8..15 (+16 for hi half)

  // staging assignment: 16B per thread, rows 64 apart
  const int sr = tid >> 2;             // 0..63
  const int sc = (tid & 3) * 8;        // 0,8,16,24 halves

  auto stage = [&](int k0, int buf) {
    { // A tile (128 x 32 halves): direct async global->LDS
      const _Float16* g = A + (m0 + sr) * (long long)lda + k0 + sc;
      async_load_b128(lds_addr_of(&As[buf][sr * LDH + sc]), g);
      async_load_b128(lds_addr_of(&As[buf][(sr + 64) * LDH + sc]),
                      g + (long long)64 * lda);
    }
    if (!B_KN) { // B tile already N x K: async path
      const _Float16* g = Bm + (n0 + sr) * (long long)ldb + k0 + sc;
      async_load_b128(lds_addr_of(&Bs[buf][sr * LDH + sc]), g);
      async_load_b128(lds_addr_of(&Bs[buf][(sr + 64) * LDH + sc]),
                      g + (long long)64 * ldb);
    } else {     // B stored K x N: transpose through registers
      const int kk = tid & 31;          // K within slab
      const int ng = (tid >> 5) * 16;   // N group base
      union { vf4 f[2]; _Float16 h[16]; } u;
      const _Float16* g = Bm + (long long)(k0 + kk) * ldb + n0 + ng;
      u.f[0] = *(const vf4*)g;
      u.f[1] = *(const vf4*)(g + 8);
#pragma unroll
      for (int i = 0; i < 16; ++i) Bs[buf][(ng + i) * LDH + kk] = u.h[i];
    }
  };

  stage(0, 0);
  wait_asynccnt0();
  __syncthreads();

  int buf = 0;
  for (int k0 = 0; k0 < K; k0 += 32) {
    const int nk = k0 + 32;
    if (nk < K) stage(nk, buf ^ 1);   // prefetch next slab into other buffer

    Frag16 fa[4], fb[2];
#pragma unroll
    for (int i = 0; i < 4; ++i) {
      const _Float16* p = &As[buf][(wm * 64 + i * 16 + arow) * LDH + ako];
      fa[i].h[0] = *(const v8h*)p;
      fa[i].h[1] = *(const v8h*)(p + 16);
    }
#pragma unroll
    for (int j = 0; j < 2; ++j) {
      const _Float16* p = &Bs[buf][(wn * 32 + j * 16 + arow) * LDH + bko];
      fb[j].h[0] = *(const v8h*)p;
      fb[j].h[1] = *(const v8h*)(p + 16);
    }
#pragma unroll
    for (int i = 0; i < 4; ++i)
#pragma unroll
      for (int j = 0; j < 2; ++j)
        acc[i][j] = __builtin_amdgcn_wmma_f32_16x16x32_f16(
            false, fa[i].v, false, fb[j].v, (short)0, acc[i][j], false, false);

    wait_asynccnt0();   // next slab landed in LDS
    __syncthreads();    // all waves done reading current slab
    buf ^= 1;
  }

  // ---- epilogue: C layout = VGPR r -> M = r + 8*(lane>=16), N = lane&15 ----
#pragma unroll
  for (int i = 0; i < 4; ++i) {
    const long long gm0 = m0 + wm * 64 + i * 16 + (lane >> 4) * 8;
#pragma unroll
    for (int j = 0; j < 2; ++j) {
      const long long gn = n0 + wn * 32 + j * 16 + (lane & 15);
      const float bv = bias ? bias[gn] : 0.0f;
#pragma unroll
      for (int r = 0; r < 8; ++r) {
        const float val = acc[i][j][r] + bv;
        const long long gm = gm0 + r;
        if (OUT_HALF) {
          Ch[(long long)bz * sC + gm * ldc + gn] = (_Float16)val;
        } else if (!SPLIT) {
          Cf[(long long)bz * sC + gm * ldc + gn] = val;
        } else {
          if (gn < splitN) Cf[gm * splitN + gn] = val;
          else             Cf2[gm * splitN + (gn - splitN)] = val;
        }
      }
    }
  }
}

// ---------------------------------------------------------------------------
// Pack kernels: fp32 -> f16
// ---------------------------------------------------------------------------
__global__ __launch_bounds__(256) void pack_x_kernel(
    const float* __restrict__ qr, const float* __restrict__ qi,
    const float* __restrict__ kr, const float* __restrict__ ki,
    const float* __restrict__ vr, const float* __restrict__ vi,
    _Float16* __restrict__ Xh)
{
  long long idx = (long long)blockIdx.x * 256 + threadIdx.x; // < 8192*3072
  long long m = idx / 3072;
  int c = (int)(idx % 3072);
  int j = c >> 9, d = c & 511;
  const float* s = (j == 0) ? qr : (j == 1) ? qi : (j == 2) ? kr
                 : (j == 3) ? ki : (j == 4) ? vr : vi;
  Xh[idx] = (_Float16)s[m * 512 + d];
}

// keep only rows [0,512)->q_r, [1024,1536)->k_r, [2048,3072)->v_r|v_i
__device__ __forceinline__ int qkv_row_map(int r) {
  return (r < 512) ? r : (r < 1024) ? r + 512 : r + 1024;
}

__global__ __launch_bounds__(256) void pack_wqkv_kernel(
    const float* __restrict__ W, const float* __restrict__ b,
    _Float16* __restrict__ Wh, float* __restrict__ bg)
{
  long long idx = (long long)blockIdx.x * 256 + threadIdx.x; // < 2048*3072
  int r = (int)(idx / 3072);
  int k = (int)(idx % 3072);
  Wh[idx] = (_Float16)W[(long long)qkv_row_map(r) * 3072 + k];
  if (idx < 2048) bg[idx] = b[qkv_row_map((int)idx)];
}

__global__ __launch_bounds__(256) void pack_h_kernel(
    const float* __restrict__ W, _Float16* __restrict__ Wh, long long n)
{
  long long idx = (long long)blockIdx.x * 256 + threadIdx.x;
  if (idx < n) Wh[idx] = (_Float16)W[idx];
}

// ---------------------------------------------------------------------------
// Row softmax with scale + key-padding mask, f32 scores -> f16 probs
// grid = (L, B); 128 threads; L = 2048 -> 16 elements/thread
// ---------------------------------------------------------------------------
__global__ __launch_bounds__(128) void softmax_kernel(
    const float* __restrict__ S, const unsigned char* __restrict__ mask,
    _Float16* __restrict__ P, float scale)
{
  const int L = 2048;
  const int tid = threadIdx.x;
  const long long base = ((long long)blockIdx.y * L + blockIdx.x) * (long long)L;
  const unsigned char* mb = mask + (long long)blockIdx.y * L;
  __shared__ float red[128];
  const float NEG_INF = -__builtin_inff();

  float vals[16];
  float mx = NEG_INF;
#pragma unroll
  for (int i = 0; i < 16; ++i) {
    int k = i * 128 + tid;
    float s = mb[k] ? NEG_INF : S[base + k] * scale;
    vals[i] = s;
    mx = fmaxf(mx, s);
  }
  red[tid] = mx; __syncthreads();
  for (int s = 64; s > 0; s >>= 1) {
    if (tid < s) red[tid] = fmaxf(red[tid], red[tid + s]);
    __syncthreads();
  }
  mx = red[0]; __syncthreads();

  const bool ok = mx > NEG_INF;  // false => entire row masked
  float sum = 0.f;
#pragma unroll
  for (int i = 0; i < 16; ++i) {
    float e = ok ? __expf(vals[i] - mx) : 0.f;
    vals[i] = e;
    sum += e;
  }
  red[tid] = sum; __syncthreads();
  for (int s = 64; s > 0; s >>= 1) {
    if (tid < s) red[tid] += red[tid + s];
    __syncthreads();
  }
  sum = red[0];
  const float inv = (sum > 0.f) ? 1.f / sum : 0.f;
#pragma unroll
  for (int i = 0; i < 16; ++i) {
    int k = i * 128 + tid;
    P[base + k] = (_Float16)(vals[i] * inv);
  }
}

// ---------------------------------------------------------------------------
extern "C" void kernel_launch(void* const* d_in, const int* in_sizes, int n_in,
                              void* d_out, int out_size, void* d_ws, size_t ws_size,
                              hipStream_t stream)
{
  (void)in_sizes; (void)n_in; (void)out_size; (void)ws_size;
  const float* q_r  = (const float*)d_in[0];
  const float* q_i  = (const float*)d_in[1];
  const float* k_r  = (const float*)d_in[2];
  const float* k_i  = (const float*)d_in[3];
  const float* v_r  = (const float*)d_in[4];
  const float* v_i  = (const float*)d_in[5];
  const unsigned char* pad = (const unsigned char*)d_in[6];
  const float* W_qkv = (const float*)d_in[7];
  const float* b_qkv = (const float*)d_in[8];
  const float* W_out = (const float*)d_in[9];
  const float* b_out = (const float*)d_in[10];
  float* out = (float*)d_out;

  // workspace carve-out (hipMalloc base is 256B aligned)
  char* ws = (char*)d_ws;
  size_t off = 0;
  auto take = [&](size_t bytes) -> char* {
    char* p = ws + off;
    off = (off + bytes + 255) & ~(size_t)255;
    return p;
  };
  _Float16* Xh   = (_Float16*)take(8192ULL * 3072 * 2);   // packed input, f16
  _Float16* Wqh  = (_Float16*)take(2048ULL * 3072 * 2);   // pruned W_qkv rows, f16
  float*    bqg  = (float*)   take(2048ULL * 4);          // gathered bias
  _Float16* QKVh = (_Float16*)take(8192ULL * 2048 * 2);   // [q_r | k_r | v_r | v_i]
  float*    S    = (float*)   take(4ULL * 2048 * 2048 * 4); // scores f32
  _Float16* P    = (_Float16*)take(4ULL * 2048 * 2048 * 2); // probs f16
  _Float16* Ori  = (_Float16*)take(8192ULL * 1024 * 2);   // attn output [real|imag]
  _Float16* Woh  = (_Float16*)take(1024ULL * 1024 * 2);   // W_out f16

  const dim3 blk(256);

  // 1) pack operands to f16
  pack_x_kernel   <<<8192 * 3072 / 256, blk, 0, stream>>>(q_r, q_i, k_r, k_i, v_r, v_i, Xh);
  pack_wqkv_kernel<<<2048 * 3072 / 256, blk, 0, stream>>>(W_qkv, b_qkv, Wqh, bqg);
  pack_h_kernel   <<<1024 * 1024 / 256, blk, 0, stream>>>(W_out, Woh, 1024LL * 1024);

  // 2) QKV projection (only live blocks): (8192x3072)*(2048x3072)^T -> f16
  gemm_wmma<false, true, false><<<dim3(2048 / 128, 8192 / 128, 1), blk, 0, stream>>>(
      Xh, Wqh, bqg, QKVh, nullptr, nullptr,
      3072, 3072, 3072, 2048, 0, 0, 0, 0);

  // 3) scores = q_r . k_r^T  (per batch: 2048x2048, K=512) -> f32
  gemm_wmma<false, false, false><<<dim3(16, 16, 4), blk, 0, stream>>>(
      QKVh, QKVh + 512, nullptr, nullptr, S, nullptr,
      512, 2048, 2048, 2048,
      2048LL * 2048, 2048LL * 2048, 2048LL * 2048, 0);

  // 4) softmax(scale, key-pad mask) -> f16 probs
  softmax_kernel<<<dim3(2048, 4), dim3(128), 0, stream>>>(S, pad, P, 0.04419417382415922f);

  // 5) attn . [v_r | v_i]  (B stored token-major K x N -> transpose path) -> f16
  gemm_wmma<true, true, false><<<dim3(8, 16, 4), blk, 0, stream>>>(
      P, QKVh + 1024, nullptr, Ori, nullptr, nullptr,
      2048, 2048, 2048, 1024,
      2048LL * 2048, 2048LL * 2048, 2048LL * 1024, 0);

  // 6) output projection + bias, split columns into (out_real, out_imag)
  gemm_wmma<false, false, true><<<dim3(8, 64, 1), blk, 0, stream>>>(
      Ori, Woh, b_out, nullptr, out, out + 8192LL * 512,
      1024, 1024, 1024, 1024, 0, 0, 0, 512);
}
